// MNT_24644522344574
// MI455X (gfx1250) — matrix operand order repaired and checked by reading, not compiled, not run on hardware
//
#include <hip/hip_runtime.h>
#include <hip/hip_bf16.h>

// ---------------------------------------------------------------------------
// Types for CDNA5 WMMA (wave32): D(f32 16x16) = A(f16 16x32) * B(f16 32x16) + C
// ---------------------------------------------------------------------------
typedef _Float16 v16h __attribute__((ext_vector_type(16)));
typedef _Float16 h8   __attribute__((ext_vector_type(8)));
typedef _Float16 h4   __attribute__((ext_vector_type(4)));
typedef float    v8f  __attribute__((ext_vector_type(8)));

union Frag { v16h v; h8 h[2]; };
union Pack2 { _Float16 h[2]; uint32_t u; };

#define M_TOK   16384      // B*N tokens
#define D_DIM   256
#define EPS_F   1e-5f

// ---------------------------------------------------------------------------
// CDNA5 async global->LDS copy (ASYNCcnt-tracked; per-lane straight copy,
// semantics per CDNA5 ISA).
// ---------------------------------------------------------------------------
__device__ __forceinline__ void async_copy_b128(uint32_t lds_off, uint64_t gaddr) {
    asm volatile("global_load_async_to_lds_b128 %0, %1, off"
                 :: "v"(lds_off), "v"(gaddr) : "memory");
}
__device__ __forceinline__ void wait_async0() {
    asm volatile("s_wait_asynccnt 0x0" ::: "memory");
}
// low 32 bits of a generic pointer to __shared__ == LDS byte offset
__device__ __forceinline__ uint32_t lds_off32(const void* p) {
    return (uint32_t)(uintptr_t)p;
}

// ---------------------------------------------------------------------------
// f32 -> f16 conversion (weights)
// ---------------------------------------------------------------------------
__global__ __launch_bounds__(256) void k_cvt_f16(const float* __restrict__ src,
                                                 _Float16* __restrict__ dst, int n) {
    int i = blockIdx.x * 256 + threadIdx.x;
    if (i < n) dst[i] = (_Float16)src[i];
}

// ---------------------------------------------------------------------------
// Sinusoidal PE table: PEtab[n][d], n=0..1023, d=0..255 (computed once)
// ---------------------------------------------------------------------------
__global__ __launch_bounds__(256) void k_pe(float* __restrict__ PEtab) {
    int n = blockIdx.x, d = threadIdx.x;
    float i2 = (float)(d >> 1);
    float ang = (float)n * powf(10000.0f, -2.0f * i2 / (float)D_DIM);
    PEtab[n * D_DIM + d] = (d & 1) ? cosf(ang) : sinf(ang);
}

// ---------------------------------------------------------------------------
// Patchify: img [16,3,512,512] -> A0 [16384, 768] f16
// ---------------------------------------------------------------------------
__global__ __launch_bounds__(256) void k_patchify(const float* __restrict__ img,
                                                  _Float16* __restrict__ A0) {
    int r  = blockIdx.x;
    int b  = r >> 10;
    int py = (r >> 5) & 31;
    int px = r & 31;
    int i  = threadIdx.x >> 4;
    int j  = threadIdx.x & 15;
    size_t rowbase = (size_t)r * 768;
    for (int c = 0; c < 3; ++c) {
        size_t idx = (((size_t)b * 3 + c) * 512 + (py * 16 + i)) * 512 + (px * 16 + j);
        A0[rowbase + c * 256 + threadIdx.x] = (_Float16)img[idx];
    }
}

// ---------------------------------------------------------------------------
// Generic WMMA GEMM, double-buffered: out = act(A@W + bias) (+ residual)
// BM=64 BN=64 BK=32, 256 threads = 8 waves. A tile staged via async
// global->LDS into alternating buffers; B tile transposed into [n][k] with
// packed b32 stores placed AFTER the WMMAs so global latency overlaps math.
// ---------------------------------------------------------------------------
#define BK 32
#define LDA 40   // halves; 80B row stride (16B multiple)
#define LDB 40

__device__ __forceinline__ float gelu_tanh(float x) {
    float x3 = x * x * x;
    return 0.5f * x * (1.0f + tanhf(0.7978845608028654f * (x + 0.044715f * x3)));
}

__global__ __launch_bounds__(256) void k_gemm(const _Float16* __restrict__ A,
                                              const _Float16* __restrict__ W,
                                              const float* __restrict__ bias,
                                              float* __restrict__ outF,
                                              _Float16* __restrict__ outH,
                                              const float* __restrict__ res,
                                              int M, int N, int K, int act) {
    __shared__ _Float16 sA[2][64 * LDA];
    __shared__ _Float16 sB[2][64 * LDB];

    int tid  = threadIdx.x;
    int lane = tid & 31;
    int wave = tid >> 5;
    int wm   = wave >> 1;        // row tile 0..3
    int wn   = wave & 1;         // col pair 0..1
    int l16  = lane & 15;
    int lh   = lane >> 4;

    int row0 = blockIdx.y * 64;
    int col0 = blockIdx.x * 64;

    // --- staging assignments (fixed per thread) ---
    int rA   = tid >> 2;
    int segA = (tid & 3) * 8;
    uint32_t ldsA0 = lds_off32(&sA[0][rA * LDA + segA]);
    uint32_t ldsA1 = lds_off32(&sA[1][rA * LDA + segA]);
    uint64_t gA    = (uint64_t)(uintptr_t)(A + (size_t)(row0 + rA) * K + segA);

    int kB = (tid >> 4) * 2;         // even k row: 0,2,...,30
    int nB = (tid & 15) * 4;         // n quad base
    const _Float16* gB = W + (size_t)kB * N + col0 + nB;

    v8f acc0 = {}; v8f acc1 = {};

    // --- prologue: stage tile 0 into buffer 0 ---
    async_copy_b128(ldsA0, gA);
    gA += (uint64_t)BK * sizeof(_Float16);
    {
        h4 b0 = *(const h4*)(gB);
        h4 b1 = *(const h4*)(gB + N);
        gB += (size_t)BK * N;
        #pragma unroll
        for (int i = 0; i < 4; ++i) {
            Pack2 p; p.h[0] = b0[i]; p.h[1] = b1[i];
            *(uint32_t*)(&sB[0][(nB + i) * LDB + kB]) = p.u;
        }
    }
    wait_async0();
    __syncthreads();

    int nk  = K / BK;
    int buf = 0;
    for (int t = 0; t < nk; ++t) {
        int  nbuf     = buf ^ 1;
        bool haveNext = (t + 1 < nk);

        // issue next tile's A (async) and B (global loads) before computing
        h4 b0 = {}; h4 b1 = {};
        if (haveNext) {
            async_copy_b128(buf ? ldsA0 : ldsA1, gA);
            gA += (uint64_t)BK * sizeof(_Float16);
            b0 = *(const h4*)(gB);
            b1 = *(const h4*)(gB + N);
            gB += (size_t)BK * N;
        }

        // --- compute tile t from sA[buf]/sB[buf] ---
        Frag a;
        {
            const _Float16* ap = &sA[buf][(wm * 16 + l16) * LDA];
            a.h[0] = *(const h8*)(ap + lh * 8);
            a.h[1] = *(const h8*)(ap + 16 + lh * 8);
        }
        #pragma unroll
        for (int cc = 0; cc < 2; ++cc) {
            Frag b;
            const _Float16* bp = &sB[buf][((wn * 2 + cc) * 16 + l16) * LDB];
            b.h[0] = *(const h8*)(bp + lh * 8);
            b.h[1] = *(const h8*)(bp + 16 + lh * 8);
            if (cc == 0)
                acc0 = __builtin_amdgcn_wmma_f32_16x16x32_f16(false, a.v, false, b.v,
                                                              (short)0, acc0, false, false);
            else
                acc1 = __builtin_amdgcn_wmma_f32_16x16x32_f16(false, a.v, false, b.v,
                                                              (short)0, acc1, false, false);
        }

        // scatter next B tile (global-load latency overlapped with WMMAs above)
        if (haveNext) {
            #pragma unroll
            for (int i = 0; i < 4; ++i) {
                Pack2 p; p.h[0] = b0[i]; p.h[1] = b1[i];
                *(uint32_t*)(&sB[nbuf][(nB + i) * LDB + kB]) = p.u;
            }
        }
        wait_async0();
        __syncthreads();
        buf = nbuf;
    }

    // --- epilogue: element (m = wm*16 + j + 8*lh, n = tile + l16)
    #pragma unroll
    for (int cc = 0; cc < 2; ++cc) {
        v8f acc = cc ? acc1 : acc0;
        int ncol = col0 + (wn * 2 + cc) * 16 + l16;
        float bv = bias[ncol];
        #pragma unroll
        for (int j = 0; j < 8; ++j) {
            int mrow = row0 + wm * 16 + j + lh * 8;
            float v = acc[j] + bv;
            if (act == 1) v = gelu_tanh(v);
            size_t o = (size_t)mrow * N + ncol;
            if (res)  v += res[o];
            if (outF) outF[o] = v;
            if (outH) outH[o] = (_Float16)v;
        }
    }
}

// ---------------------------------------------------------------------------
// LayerNorm (over D=256) + PE-table add; writes f32 (in place) + f16 copy
// ---------------------------------------------------------------------------
__global__ __launch_bounds__(256) void k_ln_pe(float* __restrict__ X,
                                               _Float16* __restrict__ Xh,
                                               const float* __restrict__ g,
                                               const float* __restrict__ b,
                                               const float* __restrict__ PEtab) {
    __shared__ float red[256];
    int row = blockIdx.x;
    int d   = threadIdx.x;
    float x = X[(size_t)row * D_DIM + d];

    red[d] = x; __syncthreads();
    for (int s = 128; s > 0; s >>= 1) { if (d < s) red[d] += red[d + s]; __syncthreads(); }
    float mu = red[0] * (1.0f / D_DIM);
    __syncthreads();

    float df = x - mu;
    red[d] = df * df; __syncthreads();
    for (int s = 128; s > 0; s >>= 1) { if (d < s) red[d] += red[d + s]; __syncthreads(); }
    float var = red[0] * (1.0f / D_DIM);

    float y = df * rsqrtf(var + EPS_F) * g[d] + b[d];
    y += PEtab[(row & 1023) * D_DIM + d];

    X[(size_t)row * D_DIM + d]  = y;
    Xh[(size_t)row * D_DIM + d] = (_Float16)y;
}

// ---------------------------------------------------------------------------
// Windowed attention: one block per (b, window, head); 128 threads = 4 waves.
// ---------------------------------------------------------------------------
#define LQK 40   // Q/K LDS stride (halves)
#define LS  68   // S  LDS stride (f32)
#define LP  72   // P / Vt LDS stride (halves, 144B)

__global__ __launch_bounds__(128) void k_attn(const _Float16* __restrict__ QKV,
                                              _Float16* __restrict__ AO) {
    __shared__ _Float16 sQ[64 * LQK];
    __shared__ _Float16 sK[64 * LQK];
    __shared__ _Float16 sVt[32 * LP];     // transposed: [dh][key]
    __shared__ float    sS[64 * LS];
    __shared__ _Float16 sP[64 * LP];

    int tid  = threadIdx.x;
    int lane = tid & 31;
    int wave = tid >> 5;           // row-tile of queries
    int l16  = lane & 15;
    int lh   = lane >> 4;

    int head = blockIdx.x & 7;
    int bw   = blockIdx.x >> 3;    // b*16 + window
    size_t tok0 = (size_t)bw * 64;

    // --- stage Q, K via async b128 copies (2 tasks each per thread)
    for (int i = tid; i < 256; i += 128) {
        int t = i >> 2, seg = (i & 3) * 8;
        size_t base = (tok0 + t) * 768 + head * 32 + seg;
        async_copy_b128(lds_off32(sQ + t * LQK + seg),
                        (uint64_t)(uintptr_t)(QKV + base));
        async_copy_b128(lds_off32(sK + t * LQK + seg),
                        (uint64_t)(uintptr_t)(QKV + base + 256));
    }
    // --- stage V transposed: sVt[d][t], token pairs packed as b32
    {
        int tp  = tid >> 2;            // token pair 0..31
        int seg = (tid & 3) * 8;       // dh segment
        size_t b0 = (tok0 + 2 * tp) * 768 + 512 + head * 32 + seg;
        h8 v0 = *(const h8*)(QKV + b0);
        h8 v1 = *(const h8*)(QKV + b0 + 768);
        #pragma unroll
        for (int i = 0; i < 8; ++i) {
            Pack2 p; p.h[0] = v0[i]; p.h[1] = v1[i];
            *(uint32_t*)(sVt + (seg + i) * LP + 2 * tp) = p.u;
        }
    }
    wait_async0();
    __syncthreads();

    // ---- S = Q K^T * 1/sqrt(dh)
    {
        Frag a;
        const _Float16* ap = sQ + (wave * 16 + l16) * LQK;
        a.h[0] = *(const h8*)(ap + lh * 8);
        a.h[1] = *(const h8*)(ap + 16 + lh * 8);
        #pragma unroll
        for (int c = 0; c < 4; ++c) {
            Frag b;
            const _Float16* bp = sK + (c * 16 + l16) * LQK;
            b.h[0] = *(const h8*)(bp + lh * 8);
            b.h[1] = *(const h8*)(bp + 16 + lh * 8);
            v8f s = {};
            s = __builtin_amdgcn_wmma_f32_16x16x32_f16(false, a.v, false, b.v,
                                                       (short)0, s, false, false);
            #pragma unroll
            for (int j = 0; j < 8; ++j)
                sS[(wave * 16 + j + lh * 8) * LS + c * 16 + l16] = s[j] * 0.17677669529663687f;
        }
    }
    __syncthreads();

    // ---- softmax over 64 keys (threads 0..63, one row each)
    if (tid < 64) {
        float mx = -1e30f;
        for (int n = 0; n < 64; ++n) mx = fmaxf(mx, sS[tid * LS + n]);
        float sum = 0.0f;
        float e[64];
        for (int n = 0; n < 64; ++n) { e[n] = expf(sS[tid * LS + n] - mx); sum += e[n]; }
        float inv = 1.0f / sum;
        for (int n = 0; n < 64; ++n) sP[tid * LP + n] = (_Float16)(e[n] * inv);
    }
    __syncthreads();

    // ---- O = P V   (K-dim = 64 keys -> two k-steps of 32)
    v8f o0 = {}; v8f o1 = {};
    #pragma unroll
    for (int s = 0; s < 2; ++s) {
        Frag a;
        const _Float16* pp = sP + (wave * 16 + l16) * LP + s * 32;
        a.h[0] = *(const h8*)(pp + lh * 8);
        a.h[1] = *(const h8*)(pp + 16 + lh * 8);
        #pragma unroll
        for (int c = 0; c < 2; ++c) {
            Frag b;
            const _Float16* vp = sVt + (c * 16 + l16) * LP + s * 32;
            b.h[0] = *(const h8*)(vp + lh * 8);
            b.h[1] = *(const h8*)(vp + 16 + lh * 8);
            if (c == 0)
                o0 = __builtin_amdgcn_wmma_f32_16x16x32_f16(false, a.v, false, b.v,
                                                            (short)0, o0, false, false);
            else
                o1 = __builtin_amdgcn_wmma_f32_16x16x32_f16(false, a.v, false, b.v,
                                                            (short)0, o1, false, false);
        }
    }
    #pragma unroll
    for (int c = 0; c < 2; ++c) {
        v8f o = c ? o1 : o0;
        #pragma unroll
        for (int j = 0; j < 8; ++j) {
            int t = wave * 16 + j + lh * 8;
            AO[(tok0 + t) * 256 + head * 32 + c * 16 + l16] = (_Float16)o[j];
        }
    }
}

// ---------------------------------------------------------------------------
// BatchNorm over (B,N): two-stage, fully coalesced, no atomics.
// Stage 1: block owns 64 contiguous rows; thread owns channel tid.
// ---------------------------------------------------------------------------
__global__ __launch_bounds__(256) void k_bn_part(const float* __restrict__ X,
                                                 float* __restrict__ p1,
                                                 float* __restrict__ p2) {
    int d  = threadIdx.x;
    int r0 = blockIdx.x * 64;
    float s = 0.0f, s2 = 0.0f;
    for (int i = 0; i < 64; ++i) {
        float v = X[(size_t)(r0 + i) * D_DIM + d];
        s += v; s2 += v * v;
    }
    p1[blockIdx.x * D_DIM + d] = s;
    p2[blockIdx.x * D_DIM + d] = s2;
}

__global__ __launch_bounds__(256) void k_bn_reduce(const float* __restrict__ p1,
                                                   const float* __restrict__ p2,
                                                   float* __restrict__ stats) {
    int d = threadIdx.x;
    float s = 0.0f, s2 = 0.0f;
    for (int i = 0; i < 256; ++i) {
        s  += p1[i * D_DIM + d];
        s2 += p2[i * D_DIM + d];
    }
    float m = s * (1.0f / M_TOK);
    stats[d]         = m;
    stats[D_DIM + d] = s2 * (1.0f / M_TOK) - m * m;
}

__global__ __launch_bounds__(256) void k_bn_apply(const float* __restrict__ X,
                                                  const float* __restrict__ stats,
                                                  const float* __restrict__ g,
                                                  const float* __restrict__ b,
                                                  float* __restrict__ outF,
                                                  _Float16* __restrict__ outH) {
    size_t idx = (size_t)blockIdx.x * 256 + threadIdx.x;
    int d = idx & (D_DIM - 1);
    float m = stats[d];
    float v = stats[D_DIM + d];
    float y = (X[idx] - m) * rsqrtf(v + EPS_F) * g[d] + b[d];
    outF[idx] = y;
    if (outH) outH[idx] = (_Float16)y;
}

// ---------------------------------------------------------------------------
// Orchestration
// ---------------------------------------------------------------------------
extern "C" void kernel_launch(void* const* d_in, const int* in_sizes, int n_in,
                              void* d_out, int out_size, void* d_ws, size_t ws_size,
                              hipStream_t stream) {
    const float* img     = (const float*)d_in[0];
    const float* patch_w = (const float*)d_in[1];
    const float* patch_b = (const float*)d_in[2];
    const float* ln_g    = (const float*)d_in[3];
    const float* ln_b    = (const float*)d_in[4];
    const float* ff_w1   = (const float*)d_in[5];
    const float* ff_b1   = (const float*)d_in[6];
    const float* ff_w2   = (const float*)d_in[7];
    const float* ff_b2   = (const float*)d_in[8];
    const float* qkv_w   = (const float*)d_in[9];
    const float* qkv_b   = (const float*)d_in[10];
    const float* proj_w  = (const float*)d_in[11];
    const float* proj_b  = (const float*)d_in[12];
    const float* bn1_g   = (const float*)d_in[13];
    const float* bn1_b   = (const float*)d_in[14];
    const float* bn2_g   = (const float*)d_in[15];
    const float* bn2_b   = (const float*)d_in[16];

    // workspace layout (bytes)
    char* w = (char*)d_ws;
    _Float16* A0   = (_Float16*)w; w += (size_t)M_TOK * 768 * 2;   // patchified; reused as QKV
    _Float16* Wp   = (_Float16*)w; w += 983040 * 2;                // f16 weight pool
    float*    X0   = (float*)w;    w += (size_t)M_TOK * 256 * 4;   // post LN+PE; reused as X5
    _Float16* X0h  = (_Float16*)w; w += (size_t)M_TOK * 256 * 2;
    _Float16* Hh   = (_Float16*)w; w += (size_t)M_TOK * 1024 * 2;  // FFN hidden (gelu'd)
    _Float16* X1h  = (_Float16*)w; w += (size_t)M_TOK * 256 * 2;   // FFN1 out (attn input)
    _Float16* AOh  = (_Float16*)w; w += (size_t)M_TOK * 256 * 2;   // attention out
    float*    X3   = (float*)w;    w += (size_t)M_TOK * 256 * 4;   // proj + X0; BN1 in-place
    _Float16* X4h  = (_Float16*)w; w += (size_t)M_TOK * 256 * 2;
    float*    PEt  = (float*)w;    w += (size_t)1024 * 256 * 4;    // PE table
    float*    bp1  = (float*)w;    w += (size_t)256 * 256 * 4;     // BN partials
    float*    bp2  = (float*)w;    w += (size_t)256 * 256 * 4;
    float*    stats= (float*)w;    w += 4096;

    _Float16* pwh   = Wp;            // 768*256
    _Float16* w1h   = Wp + 196608;   // 256*1024
    _Float16* w2h   = Wp + 458752;   // 1024*256
    _Float16* qkvh  = Wp + 720896;   // 256*768
    _Float16* projh = Wp + 917504;   // 256*256

    // 1) weights -> f16 ; PE table
    k_cvt_f16<<<(196608 + 255) / 256, 256, 0, stream>>>(patch_w, pwh, 196608);
    k_cvt_f16<<<(262144 + 255) / 256, 256, 0, stream>>>(ff_w1, w1h, 262144);
    k_cvt_f16<<<(262144 + 255) / 256, 256, 0, stream>>>(ff_w2, w2h, 262144);
    k_cvt_f16<<<(196608 + 255) / 256, 256, 0, stream>>>(qkv_w, qkvh, 196608);
    k_cvt_f16<<<( 65536 + 255) / 256, 256, 0, stream>>>(proj_w, projh, 65536);
    k_pe<<<1024, 256, 0, stream>>>(PEt);

    // 2) patchify
    k_patchify<<<M_TOK, 256, 0, stream>>>(img, A0);

    // 3) patch linear -> X0 f32
    k_gemm<<<dim3(256 / 64, M_TOK / 64), 256, 0, stream>>>(
        A0, pwh, patch_b, X0, nullptr, nullptr, M_TOK, 256, 768, 0);

    // 4) LN + PE -> X0 (f32, in place) + X0h (f16)
    k_ln_pe<<<M_TOK, 256, 0, stream>>>(X0, X0h, ln_g, ln_b, PEt);

    // 5) FFN1: gelu(X0 @ w1 + b1) -> Hh ; Hh @ w2 + b2 -> X1h
    k_gemm<<<dim3(1024 / 64, M_TOK / 64), 256, 0, stream>>>(
        X0h, w1h, ff_b1, nullptr, Hh, nullptr, M_TOK, 1024, 256, 1);
    k_gemm<<<dim3(256 / 64, M_TOK / 64), 256, 0, stream>>>(
        Hh, w2h, ff_b2, nullptr, X1h, nullptr, M_TOK, 256, 1024, 0);

    // 6) QKV (reuse A0 buffer)
    k_gemm<<<dim3(768 / 64, M_TOK / 64), 256, 0, stream>>>(
        X1h, qkvh, qkv_b, nullptr, A0, nullptr, M_TOK, 768, 256, 0);

    // 7) windowed attention: 16 B * 16 windows * 8 heads blocks
    k_attn<<<16 * 16 * 8, 128, 0, stream>>>(A0, AOh);

    // 8) proj + residual(X0) -> X3 f32
    k_gemm<<<dim3(256 / 64, M_TOK / 64), 256, 0, stream>>>(
        AOh, projh, proj_b, X3, nullptr, X0, M_TOK, 256, 256, 0);

    // 9) BN1 -> X3 (in place) + X4h f16
    k_bn_part<<<M_TOK / 64, 256, 0, stream>>>(X3, bp1, bp2);
    k_bn_reduce<<<1, 256, 0, stream>>>(bp1, bp2, stats);
    k_bn_apply<<<M_TOK, 256, 0, stream>>>(X3, stats, bn1_g, bn1_b, X3, X4h);

    // 10) FFN2 + residual(X4=X3) -> X5 (reuse X0 buffer)
    k_gemm<<<dim3(1024 / 64, M_TOK / 64), 256, 0, stream>>>(
        X4h, w1h, ff_b1, nullptr, Hh, nullptr, M_TOK, 1024, 256, 1);
    k_gemm<<<dim3(256 / 64, M_TOK / 64), 256, 0, stream>>>(
        Hh, w2h, ff_b2, X0, nullptr, X3, M_TOK, 256, 1024, 0);

    // 11) BN2 -> d_out (f32)
    k_bn_part<<<M_TOK / 64, 256, 0, stream>>>(X0, bp1, bp2);
    k_bn_reduce<<<1, 256, 0, stream>>>(bp1, bp2, stats);
    k_bn_apply<<<M_TOK, 256, 0, stream>>>(X0, stats, bn2_g, bn2_b, (float*)d_out, nullptr);
}
